// Encoder_29755533427524
// MI455X (gfx1250) — compile-verified, matrix operand order checked by reading, not run on hardware
//
#include <hip/hip_runtime.h>
#include <hip/hip_bf16.h>
#include <cstdint>

// ---------------- problem constants (from reference) ----------------
#define EE 256      // embedding dim
#define HH 256      // hidden dim
#define LL 128      // seq len
#define NN 1024     // (N_UTT-1)*B sequences
#define BB 256      // batch
#define UU 4        // utterances per item (N_UTT-1)
#define G4H (4*HH)  // 1024 gate columns
#define WLROW (5*HH) // (MAX_UTT-1)*H = 1280, Wl row length
#define MT 32       // sequences (rows) per workgroup in the LSTM kernel

typedef __attribute__((ext_vector_type(16))) __bf16 v16bf;
typedef __attribute__((ext_vector_type(8)))  float  v8f;
typedef __attribute__((ext_vector_type(4)))  int    i4; // trivially-constructible 16B mover

// address-space-qualified views: force global_load_b128 (not flat) + async-to-LDS operands
typedef __attribute__((address_space(1))) i4     a1_i4;
typedef __attribute__((address_space(3))) i4     a3_i4;
typedef __attribute__((address_space(1))) __bf16 a1_bf;

union V16BF { v16bf v; i4 q[2]; };

#if __has_builtin(__builtin_amdgcn_global_load_async_to_lds_b128) && __has_builtin(__builtin_amdgcn_s_wait_asynccnt)
#define USE_ASYNC_LDS 1
#else
#define USE_ASYNC_LDS 0
#endif

__device__ __forceinline__ float sigmoid_f(float x) { return 1.0f / (1.0f + __expf(-x)); }

// ---------------- k0: f32 -> bf16 weight conversion ----------------
__global__ void f32_to_bf16_kernel(const float* __restrict__ x, __bf16* __restrict__ y, int n) {
    int i = blockIdx.x * blockDim.x + threadIdx.x;
    if (i < n) y[i] = (__bf16)x[i];
}

// ---------------- k1: embedding gather (+bf16 store) and per-seq sum ----------------
__global__ void embed_gather_kernel(const int* __restrict__ tok, const float* __restrict__ emb,
                                    __bf16* __restrict__ X, float* __restrict__ se) {
    const int n = blockIdx.x;
    const int e = threadIdx.x;
    const int* trow = tok + n * LL;
    float s = 0.0f;
    for (int l = 0; l < LL; ++l) {
        int t = trow[l];
        if (l + 1 < LL)
            __builtin_prefetch(emb + (size_t)trow[l + 1] * EE + e, 0, 0); // global_prefetch_b8
        float v = emb[(size_t)t * EE + e];
        X[((size_t)n * LL + l) * EE + e] = (__bf16)v;
        s += v;
    }
    se[n * EE + e] = s;
}

// ---------------- k2: cosine vs last utt + softmax over U ----------------
__global__ void cos_softmax_kernel(const float* __restrict__ se, float* __restrict__ weight) {
    const int b = blockIdx.x;
    const int lane = threadIdx.x;
    const float* last = se + (size_t)(b * UU + UU - 1) * EE;
    float dot[UU], nrm[UU];
    for (int u = 0; u < UU; ++u) {
        const float* row = se + (size_t)(b * UU + u) * EE;
        float d = 0.0f, q = 0.0f;
        for (int e = lane; e < EE; e += 32) { float a = row[e]; d += a * last[e]; q += a * a; }
        for (int off = 16; off; off >>= 1) {
            d += __shfl_xor(d, off, 32);
            q += __shfl_xor(q, off, 32);
        }
        dot[u] = d; nrm[u] = sqrtf(q);
    }
    if (lane == 0) {
        float nl = fmaxf(nrm[UU - 1], 1e-8f);
        float c[UU], mx = -1e30f;
        for (int u = 0; u < UU; ++u) { c[u] = dot[u] / (fmaxf(nrm[u], 1e-8f) * nl); mx = fmaxf(mx, c[u]); }
        float sum = 0.0f;
        for (int u = 0; u < UU; ++u) { c[u] = __expf(c[u] - mx); sum += c[u]; }
        for (int u = 0; u < UU; ++u) weight[b * UU + u] = c[u] / sum;
    }
}

// ---------------- k3: WMMA LSTM ----------------
// One WG = MT(=32) sequences, 512 threads = 16 waves.
// Wave w owns h-columns [w*16, w*16+16): it computes the i/f/g/o gate tiles for those
// columns (tiles at gate columns g*256 + w*16), so the c/h update is wave-local and the
// c/h state lives in VGPRs. x_t tiles are double-buffered in LDS via async-to-LDS loads
// overlapped with the WMMA GEMM of the current step. Weights stream from L2 each step
// (laundered addrspace(1) pointers -> global_load_b128, not hoistable/spillable).
__global__ __launch_bounds__(512, 1)
void lstm_wmma_kernel(const __bf16* __restrict__ X, const int* __restrict__ lens, int Lseq,
                      const __bf16* __restrict__ Wi, const __bf16* __restrict__ Wh,
                      const float* __restrict__ bias,
                      float* __restrict__ out, __bf16* hTout, __bf16* cTout) {
    __shared__ __align__(16) __bf16 sX[2][MT * EE]; // double-buffered x_t tile, 2x16KB
    __shared__ __align__(16) __bf16 sH[MT * HH];    // h tile bf16 (A operand), 16KB
    __shared__ int sLen[MT];

    const int tid  = threadIdx.x;
    const int wave = tid >> 5;
    const int lane = tid & 31;
    const int n0   = blockIdx.x * MT;

    // A/B per-lane addressing for the 16-bit 16x32 operand layout:
    // lanes 0-15 hold K{0..7,16..23}, lanes 16-31 hold K{8..15,24..31} of row (lane&15).
    const int arow = lane & 15;
    const int asel = lane >> 4;
    const int hcol = wave * 16 + arow;  // this lane's h-column in gate/update phase

    for (int i = tid; i < MT * HH; i += 512) sH[i] = (__bf16)0.0f;
    if (tid < MT) sLen[tid] = lens ? lens[n0 + tid] : Lseq;

    // per-lane f32 state: rows m*16 + (r or r+8), column hcol
    float creg[2][8], hreg[2][8];
#pragma unroll
    for (int m = 0; m < 2; ++m)
#pragma unroll
        for (int r = 0; r < 8; ++r) { creg[m][r] = 0.0f; hreg[m][r] = 0.0f; }

    float bgate[4];
#pragma unroll
    for (int g = 0; g < 4; ++g) bgate[g] = bias[g * HH + hcol];

    // stage x_t tile (MT rows x 512B) for step t into buffer buf
    auto stage = [&](int t, int buf) {
        for (int i = tid; i < MT * 32; i += 512) {
            const int r  = i >> 5;   // row 0..MT-1
            const int ch = i & 31;   // 16B chunk
            const a1_bf* src = (const a1_bf*)(X + ((size_t)(n0 + r) * Lseq + t) * EE + ch * 8);
            __bf16* dst = &sX[buf][r * EE + ch * 8];
#if USE_ASYNC_LDS
            __builtin_amdgcn_global_load_async_to_lds_b128((a1_i4*)src, (a3_i4*)dst, 0, 0);
#else
            *(i4*)dst = *(const a1_i4*)src;
#endif
        }
    };

    stage(0, 0); // prologue: prefetch first step

    for (int t = 0; t < Lseq; ++t) {
        const int cur = t & 1;
#if USE_ASYNC_LDS
        __builtin_amdgcn_s_wait_asynccnt(0);   // this thread's staged loads done
#endif
        __syncthreads();                        // staged buf + sH(t) visible to all

        if (t + 1 < Lseq) stage(t + 1, cur ^ 1); // overlap next fetch with this GEMM

        // launder weight pointers (addrspace(1)-qualified) so LICM cannot hoist the
        // t-invariant weight loads out of the time loop and spill them to scratch.
        const a1_bf* wi = (const a1_bf*)Wi;
        const a1_bf* wh = (const a1_bf*)Wh;
        asm volatile("" : "+v"(wi), "+v"(wh));

        // ---- WMMA: acc[m][g] = Xt*Wi^T + H*Wh^T for gate tiles (cols g*256+wave*16) ----
        v8f acc[2][4];
        const v8f vz = {0, 0, 0, 0, 0, 0, 0, 0};
#pragma unroll
        for (int m = 0; m < 2; ++m)
#pragma unroll
            for (int g = 0; g < 4; ++g) acc[m][g] = vz;

#pragma unroll
        for (int kc = 0; kc < 8; ++kc) {
            const int kb = kc * 32;
            V16BF ax[2], ah[2];
#pragma unroll
            for (int m = 0; m < 2; ++m) {
                const __bf16* xp = &sX[cur][(m * 16 + arow) * EE + kb + asel * 8];
                ax[m].q[0] = *(const i4*)xp;  ax[m].q[1] = *(const i4*)(xp + 16);
                const __bf16* hp = &sH[(m * 16 + arow) * HH + kb + asel * 8];
                ah[m].q[0] = *(const i4*)hp;  ah[m].q[1] = *(const i4*)(hp + 16);
            }
#pragma unroll
            for (int g = 0; g < 4; ++g) {
                const int nb = g * HH + wave * 16;   // gate-column tile base
                V16BF bi, bh;
                const a1_bf* wip = wi + (size_t)(nb + arow) * EE + kb + asel * 8;
                bi.q[0] = *(const a1_i4*)wip; bi.q[1] = *(const a1_i4*)(wip + 16);
                const a1_bf* whp = wh + (size_t)(nb + arow) * HH + kb + asel * 8;
                bh.q[0] = *(const a1_i4*)whp; bh.q[1] = *(const a1_i4*)(whp + 16);
#pragma unroll
                for (int m = 0; m < 2; ++m) {
                    acc[m][g] = __builtin_amdgcn_wmma_f32_16x16x32_bf16(
                        false, ax[m].v, false, bi.v, (short)0, acc[m][g], false, false);
                    acc[m][g] = __builtin_amdgcn_wmma_f32_16x16x32_bf16(
                        false, ah[m].v, false, bh.v, (short)0, acc[m][g], false, false);
                }
            }
        }
        __syncthreads();   // all waves done reading sH before anyone rewrites it

        // ---- wave-local gate nonlinearity + masked state update (C/D layout:
        //      VGPR r -> row r (lanes 0-15) or r+8 (lanes 16-31), col = hcol) ----
#pragma unroll
        for (int m = 0; m < 2; ++m) {
#pragma unroll
            for (int r = 0; r < 8; ++r) {
                const int row = m * 16 + ((lane < 16) ? r : (r + 8));
                const int n   = n0 + row;
                const float gi = sigmoid_f(acc[m][0][r] + bgate[0]);
                const float gf = sigmoid_f(acc[m][1][r] + bgate[1]);
                const float gg = tanhf    (acc[m][2][r] + bgate[2]);
                const float go = sigmoid_f(acc[m][3][r] + bgate[3]);
                const float cold = creg[m][r];
                const float hold = hreg[m][r];
                const float cnew = gf * cold + gi * gg;
                const float hnew = go * tanhf(cnew);
                const bool active = t < sLen[row];
                const float c2 = active ? cnew : cold;
                const float h2 = active ? hnew : hold;
                creg[m][r] = c2;
                hreg[m][r] = h2;
                sH[row * HH + hcol] = (__bf16)h2;
                out[((size_t)n * Lseq + t) * HH + hcol] = active ? hnew : 0.0f;
            }
        }
    }

    if (hTout) {
#pragma unroll
        for (int m = 0; m < 2; ++m) {
#pragma unroll
            for (int r = 0; r < 8; ++r) {
                const int row = m * 16 + ((lane < 16) ? r : (r + 8));
                hTout[(size_t)(n0 + row) * HH + hcol] = (__bf16)hreg[m][r];
                cTout[(size_t)(n0 + row) * HH + hcol] = (__bf16)creg[m][r];
            }
        }
    }
}

// ---------------- k5: weighted projection: y[b,:] = (outs[b]*w[b]).flat @ Wl^T + bl ----------------
__global__ void proj_kernel(const float* __restrict__ outs, const float* __restrict__ weight,
                            const float* __restrict__ Wl, const float* __restrict__ bl,
                            float* __restrict__ y) {
    __shared__ float sx[UU * HH];
    const int b = blockIdx.x, j = threadIdx.x; // 256 threads
    for (int i = j; i < UU * HH; i += 256) {
        const int u = i / HH;
        sx[i] = outs[(size_t)(b * UU) * HH + i] * weight[b * UU + u];
    }
    __syncthreads();
    float acc = bl[j];
    const float* wrow = Wl + (size_t)j * WLROW; // padded cols beyond UU*HH multiply zeros -> skipped
#pragma unroll 4
    for (int i = 0; i < UU * HH; ++i) acc += sx[i] * wrow[i];
    y[b * HH + j] = acc;
}

// ---------------- host ----------------
extern "C" void kernel_launch(void* const* d_in, const int* in_sizes, int n_in,
                              void* d_out, int out_size, void* d_ws, size_t ws_size,
                              hipStream_t stream) {
    (void)in_sizes; (void)n_in; (void)out_size; (void)ws_size;

    const int*   tok   = (const int*)d_in[0];
    const int*   lens  = (const int*)d_in[1];
    const float* emb   = (const float*)d_in[7];
    const float* Wih   = (const float*)d_in[8];
    const float* Whh   = (const float*)d_in[9];
    const float* b     = (const float*)d_in[10];
    const float* Wih_h = (const float*)d_in[11];
    const float* Whh_h = (const float*)d_in[12];
    const float* b_h   = (const float*)d_in[13];
    const float* Wih_c = (const float*)d_in[14];
    const float* Whh_c = (const float*)d_in[15];
    const float* b_c   = (const float*)d_in[16];
    const float* Wlh   = (const float*)d_in[17];
    const float* blh   = (const float*)d_in[18];
    const float* Wlc   = (const float*)d_in[19];
    const float* blc   = (const float*)d_in[20];

    char* ws = (char*)d_ws;
    size_t off = 0;
    auto alloc = [&](size_t bytes) { void* p = ws + off; off += (bytes + 255) & ~(size_t)255; return p; };

    __bf16* Xbf   = (__bf16*)alloc((size_t)NN * LL * EE * 2);
    float*  se    = (float*) alloc((size_t)NN * EE * 4);
    float*  wgt   = (float*) alloc((size_t)BB * UU * 4);
    __bf16* WiB   = (__bf16*)alloc((size_t)G4H * EE * 2);
    __bf16* WhB   = (__bf16*)alloc((size_t)G4H * HH * 2);
    __bf16* WihB  = (__bf16*)alloc((size_t)G4H * HH * 2);
    __bf16* WhhB  = (__bf16*)alloc((size_t)G4H * HH * 2);
    __bf16* WicB  = (__bf16*)alloc((size_t)G4H * HH * 2);
    __bf16* WhcB  = (__bf16*)alloc((size_t)G4H * HH * 2);
    __bf16* hTb   = (__bf16*)alloc((size_t)NN * HH * 2);
    __bf16* cTb   = (__bf16*)alloc((size_t)NN * HH * 2);
    float*  outsH = (float*) alloc((size_t)BB * UU * HH * 4);
    float*  outsC = (float*) alloc((size_t)BB * UU * HH * 4);

    const int wn = G4H * EE;
    const int cb = (wn + 255) / 256;
    f32_to_bf16_kernel<<<cb, 256, 0, stream>>>(Wih,   WiB,  wn);
    f32_to_bf16_kernel<<<cb, 256, 0, stream>>>(Whh,   WhB,  wn);
    f32_to_bf16_kernel<<<cb, 256, 0, stream>>>(Wih_h, WihB, wn);
    f32_to_bf16_kernel<<<cb, 256, 0, stream>>>(Whh_h, WhhB, wn);
    f32_to_bf16_kernel<<<cb, 256, 0, stream>>>(Wih_c, WicB, wn);
    f32_to_bf16_kernel<<<cb, 256, 0, stream>>>(Whh_c, WhcB, wn);

    embed_gather_kernel<<<NN, EE, 0, stream>>>(tok, emb, Xbf, se);
    cos_softmax_kernel<<<BB, 32, 0, stream>>>(se, wgt);

    float* enc    = (float*)d_out;
    float* h_last = enc + (size_t)NN * LL * HH;
    float* c_last = h_last + (size_t)BB * HH;

    // main LSTM over all 1024 sequences, 128 steps
    lstm_wmma_kernel<<<NN / MT, 512, 0, stream>>>(Xbf, lens, LL, WiB, WhB, b, enc, hTb, cTb);
    // branch LSTMs over final states (256 sequences of length 4, full lens)
    lstm_wmma_kernel<<<BB / MT, 512, 0, stream>>>(hTb, nullptr, UU, WihB, WhhB, b_h, outsH, nullptr, nullptr);
    lstm_wmma_kernel<<<BB / MT, 512, 0, stream>>>(cTb, nullptr, UU, WicB, WhcB, b_c, outsC, nullptr, nullptr);

    proj_kernel<<<BB, HH, 0, stream>>>(outsH, wgt, Wlh, blh, h_last);
    proj_kernel<<<BB, HH, 0, stream>>>(outsC, wgt, Wlc, blc, c_last);
}
